// AlbertSeq2Seq_62466004353373
// MI455X (gfx1250) — compile-verified, matrix operand order checked by reading, not run on hardware
//
#include <hip/hip_runtime.h>
#include <hip/hip_bf16.h>

typedef __attribute__((ext_vector_type(16))) __bf16 v16bf;
typedef __attribute__((ext_vector_type(8)))  __bf16 bf16x8;
typedef __attribute__((ext_vector_type(8)))  float  v8f;

#define HID 1024
#define HD  64
#define NH  16
#define LDP 40   // padded LDS row stride in bf16 elements (80B = 20 banks, conflict-free, 16B aligned)

// ---------------------------------------------------------------------------
// A/B 16-bit fragment K-mapping (ISA 7.12.2): kbase = 8*(lane>>4);
// element e -> K = kbase + e (e<8) or kbase + e + 8 (e>=8):
// two contiguous 8-element (16B) runs at [kbase] and [kbase+16].
// ---------------------------------------------------------------------------
__device__ __forceinline__ v16bf fragload(const __bf16* p) {
    bf16x8 x0 = *(const bf16x8*)(p);
    bf16x8 x1 = *(const bf16x8*)(p + 16);
    v16bf r;
#pragma unroll
    for (int i = 0; i < 8; ++i) { r[i] = x0[i]; r[i + 8] = x1[i]; }
    return r;
}

__device__ __forceinline__ bf16x8 cvt8(float4 a, float4 b) {
    bf16x8 t;
    t[0] = (__bf16)a.x; t[1] = (__bf16)a.y; t[2] = (__bf16)a.z; t[3] = (__bf16)a.w;
    t[4] = (__bf16)b.x; t[5] = (__bf16)b.y; t[6] = (__bf16)b.z; t[7] = (__bf16)b.w;
    return t;
}

__device__ __forceinline__ v8f wmma_bf16(v16bf a, v16bf b, v8f c) {
    return __builtin_amdgcn_wmma_f32_16x16x32_bf16(false, a, false, b, (short)0, c,
                                                   false, false);
}

// async global->LDS 16B transfer; lds32 = wave-relative LDS byte address
__device__ __forceinline__ void async_b128(unsigned lds32, const void* gaddr) {
    asm volatile("global_load_async_to_lds_b128 %0, %1, off"
                 :: "v"(lds32), "v"((unsigned long long)(uintptr_t)gaddr)
                 : "memory");
}
#define WAIT_ASYNC_4() asm volatile("s_wait_asynccnt 0x4" ::: "memory")
#define WAIT_ASYNC_0() asm volatile("s_wait_asynccnt 0x0" ::: "memory")

// ---------------------------------------------------------------------------
// f32 -> bf16 bulk convert (done once per operand, outside all hot loops)
// ---------------------------------------------------------------------------
__global__ __launch_bounds__(256) void cvtbf_kernel(const float* __restrict__ in,
                                                    __bf16* __restrict__ out, int n8) {
    int i = blockIdx.x * blockDim.x + threadIdx.x;
    if (i < n8) {
        float4 a = ((const float4*)in)[2 * i];
        float4 b = ((const float4*)in)[2 * i + 1];
        ((bf16x8*)out)[i] = cvt8(a, b);
    }
}

// ---------------------------------------------------------------------------
// Linear: C = A[M,K](bf16) @ W[N,K](bf16)^T + bias.  Optional f32/bf16 outputs.
// 128x128 block tile, BK=32; tiles staged by GLOBAL_LOAD_ASYNC_TO_LDS_B128 into
// double-buffered padded LDS; ASYNCcnt pipeline one k-step ahead; last k-step
// peeled so the steady-state loop is branch-free.
// ---------------------------------------------------------------------------
__global__ __launch_bounds__(256) void linear_kernel(
    const __bf16* __restrict__ A, const __bf16* __restrict__ W,
    const float* __restrict__ bias, float* __restrict__ Cf, __bf16* __restrict__ Cb,
    int M, int N, int K) {
    const int tid  = threadIdx.x;
    const int lane = tid & 31;
    const int warp = tid >> 5;
    const int wRow = warp >> 1;
    const int wCol = warp & 1;
    const int r    = lane & 15;
    const int hi   = lane >> 4;
    const int kb   = hi * 8;
    const int blockM = blockIdx.y * 128;
    const int blockN = blockIdx.x * 128;

    __shared__ __attribute__((aligned(16))) __bf16 sA[2][128 * LDP];
    __shared__ __attribute__((aligned(16))) __bf16 sB[2][128 * LDP];

    // async chunk geometry: per array, 512 16B chunks = (row 0..127) x (q 0..3);
    // thread owns (row0, q) and (row0+64, q) in each of A and B.
    const int qv   = tid & 3;
    const int row0 = tid >> 2;                 // 0..63
    const __bf16* aG = A + (size_t)(blockM + row0) * K + qv * 8;
    const __bf16* wG = W + (size_t)(blockN + row0) * K + qv * 8;
    const unsigned ldsA = (unsigned)(uintptr_t)&sA[0][row0 * LDP + qv * 8];
    const unsigned ldsB = (unsigned)(uintptr_t)&sB[0][row0 * LDP + qv * 8];
    const unsigned rowHop = 64 * LDP * 2;
    const unsigned bufHop = 128 * LDP * 2;
    const size_t   gHop   = (size_t)64 * K;

    auto issue = [&](int buf, int k0) {
        const unsigned bo = buf ? bufHop : 0u;
        async_b128(ldsA + bo,          aG + k0);
        async_b128(ldsA + bo + rowHop, aG + k0 + gHop);
        async_b128(ldsB + bo,          wG + k0);
        async_b128(ldsB + bo + rowHop, wG + k0 + gHop);
    };

    v8f acc[2][4] = {};
    auto compute = [&](int buf) {
        v16bf af0 = fragload(&sA[buf][(wRow * 32 + r) * LDP + kb]);
        v16bf af1 = fragload(&sA[buf][(wRow * 32 + 16 + r) * LDP + kb]);
#pragma unroll
        for (int t = 0; t < 4; ++t) {
            v16bf bfr = fragload(&sB[buf][(wCol * 64 + t * 16 + r) * LDP + kb]);
            acc[0][t] = wmma_bf16(af0, bfr, acc[0][t]);
            acc[1][t] = wmma_bf16(af1, bfr, acc[1][t]);
        }
    };

    issue(0, 0);
    int buf = 0;
    for (int k0 = 0; k0 < K - 32; k0 += 32) {      // branch-free steady state
        issue(buf ^ 1, k0 + 32);
        WAIT_ASYNC_4();                             // this k-step's tile complete
        __syncthreads();
        compute(buf);
        __syncthreads();
        buf ^= 1;
    }
    WAIT_ASYNC_0();                                 // peeled final k-step
    __syncthreads();
    compute(buf);

    if (Cf) {
#pragma unroll
        for (int sub = 0; sub < 2; ++sub)
#pragma unroll
            for (int t = 0; t < 4; ++t) {
                const int col = blockN + wCol * 64 + t * 16 + r;
                const float bvv = bias[col];
#pragma unroll
                for (int v = 0; v < 8; ++v) {
                    const int m = blockM + wRow * 32 + sub * 16 + v + 8 * hi;
                    Cf[(size_t)m * N + col] = acc[sub][t][v] + bvv;
                }
            }
    }
    if (Cb) {
#pragma unroll
        for (int sub = 0; sub < 2; ++sub)
#pragma unroll
            for (int t = 0; t < 4; ++t) {
                const int col = blockN + wCol * 64 + t * 16 + r;
                const float bvv = bias[col];
#pragma unroll
                for (int v = 0; v < 8; ++v) {
                    const int m = blockM + wRow * 32 + sub * 16 + v + 8 * hi;
                    Cb[(size_t)m * N + col] = (__bf16)(acc[sub][t][v] + bvv);
                }
            }
    }
}

// ---------------------------------------------------------------------------
// Fused flash attention, all-bf16 operands: grid = (T/64, NH, B), 128 thr = 4 waves.
// K chunks (64s x 64d) async-to-LDS, V chunks transposed via VGPRs; both LDS
// double-buffered so chunk i+1 loads overlap chunk i compute; last chunk peeled.
// ---------------------------------------------------------------------------
__global__ __launch_bounds__(128) void attn_kernel(
    const __bf16* __restrict__ Q, const __bf16* __restrict__ Kt,
    const __bf16* __restrict__ V, const float* __restrict__ mask,
    __bf16* __restrict__ O) {
    const int lane = threadIdx.x & 31;
    const int warp = threadIdx.x >> 5;
    const int b = blockIdx.z, h = blockIdx.y;
    const int t0 = blockIdx.x * 64 + warp * 16;
    const int r  = lane & 15;
    const int hi = lane >> 4;
    const int kb = hi * 8;
    const int hoff = h * HD;

    __shared__ __attribute__((aligned(16))) __bf16 lK[2][2][64 * LDP];   // [buf][dHalf][s][LDP]
    __shared__ __attribute__((aligned(16))) __bf16 lVt[2][2][64 * LDP];  // [buf][sHalf][d][LDP]
    __shared__ __attribute__((aligned(16))) __bf16 lP[4][2][16 * LDP];

    v16bf qf[2];
    {
        const __bf16* qp = Q + (size_t)(b * 1024 + t0 + r) * HID + hoff + kb;
        qf[0] = fragload(qp);
        qf[1] = fragload(qp + 32);
    }

    v8f acc[4] = {};
    float mRow[8], lRow[8];
#pragma unroll
    for (int v = 0; v < 8; ++v) { mRow[v] = -3.0e38f; lRow[v] = 0.f; }

    // async K chunk geometry: 512 16B chunks = (dHalf, srow 0..63, q 0..3); 4/thread
    unsigned kLds[4];
    size_t   kOfs[4];
#pragma unroll
    for (int i = 0; i < 4; ++i) {
        const int c = threadIdx.x + i * 128;
        const int q = c & 3, srow = (c >> 2) & 63, dh = c >> 8;
        kLds[i] = (unsigned)(uintptr_t)&lK[0][dh][srow * LDP + q * 8];
        kOfs[i] = (size_t)srow * HID + dh * 32 + q * 8;
    }
    const unsigned kBufHop = 2 * 64 * LDP * 2;     // lK buffer-1 byte offset
    const __bf16* kBase = Kt + (size_t)(b * 1024) * HID + hoff;

    // V staging: thread stages 32 d-elements of one s-row into transposed layout
    const int si = threadIdx.x >> 1;
    const int d0 = (threadIdx.x & 1) * 32;
    const int sH = si >> 5, sc_ = si & 31;
    bf16x8 vv[4];

    auto issueK = [&](int buf, int s0) {
        const unsigned bo = buf ? kBufHop : 0u;
#pragma unroll
        for (int i = 0; i < 4; ++i)
            async_b128(kLds[i] + bo, kBase + (size_t)s0 * HID + kOfs[i]);
    };
    auto loadV = [&](int s0) {
        const __bf16* vp = V + (size_t)(b * 1024 + s0 + si) * HID + hoff + d0;
#pragma unroll
        for (int i = 0; i < 4; ++i) vv[i] = ((const bf16x8*)vp)[i];
    };
    auto storeV = [&](int buf) {
#pragma unroll
        for (int j = 0; j < 4; ++j)
#pragma unroll
            for (int e = 0; e < 8; ++e)
                lVt[buf][sH][(d0 + j * 8 + e) * LDP + sc_] = vv[j][e];
    };

    auto computeChunk = [&](int s0, int buf) {
        v8f sc[4] = {};
#pragma unroll
        for (int ts = 0; ts < 4; ++ts)
#pragma unroll
            for (int ks = 0; ks < 2; ++ks) {
                v16bf bfK = fragload(&lK[buf][ks][(ts * 16 + r) * LDP + kb]);
                sc[ts] = wmma_bf16(qf[ks], bfK, sc[ts]);
            }
        float msk[4];
#pragma unroll
        for (int ts = 0; ts < 4; ++ts) msk[ts] = mask[b * 1024 + s0 + ts * 16 + r];
        float mNew[8];
#pragma unroll
        for (int v = 0; v < 8; ++v) {
            float mv = -3.0e38f;
#pragma unroll
            for (int ts = 0; ts < 4; ++ts) {
                sc[ts][v] = sc[ts][v] * 0.125f + msk[ts];
                mv = fmaxf(mv, sc[ts][v]);
            }
            mNew[v] = fmaxf(mRow[v], mv);
        }
#pragma unroll
        for (int off = 1; off < 16; off <<= 1)
#pragma unroll
            for (int v = 0; v < 8; ++v)
                mNew[v] = fmaxf(mNew[v], __shfl_xor(mNew[v], off));
#pragma unroll
        for (int v = 0; v < 8; ++v) {
            const float alpha = __expf(mRow[v] - mNew[v]);
            mRow[v] = mNew[v];
            lRow[v] *= alpha;
#pragma unroll
            for (int d = 0; d < 4; ++d) acc[d][v] *= alpha;
#pragma unroll
            for (int ts = 0; ts < 4; ++ts) {
                const float p = __expf(sc[ts][v] - mNew[v]);
                lRow[v] += p;
                lP[warp][ts >> 1][(v + 8 * hi) * LDP + (ts & 1) * 16 + r] = (__bf16)p;
            }
        }
        v16bf pf0 = fragload(&lP[warp][0][r * LDP + kb]);
        v16bf pf1 = fragload(&lP[warp][1][r * LDP + kb]);
#pragma unroll
        for (int d = 0; d < 4; ++d) {
            v16bf vf0 = fragload(&lVt[buf][0][(d * 16 + r) * LDP + kb]);
            acc[d] = wmma_bf16(pf0, vf0, acc[d]);
            v16bf vf1 = fragload(&lVt[buf][1][(d * 16 + r) * LDP + kb]);
            acc[d] = wmma_bf16(pf1, vf1, acc[d]);
        }
    };

    issueK(0, 0);
    loadV(0);
    int buf = 0;
    for (int s0 = 0; s0 < 1024 - 64; s0 += 64) {   // branch-free steady state
        storeV(buf);
        issueK(buf ^ 1, s0 + 64);
        loadV(s0 + 64);
        WAIT_ASYNC_4();                             // current chunk's K complete
        __syncthreads();
        computeChunk(s0, buf);
        __syncthreads();
        buf ^= 1;
    }
    storeV(buf);                                    // peeled final chunk
    WAIT_ASYNC_0();
    __syncthreads();
    computeChunk(1024 - 64, buf);

#pragma unroll
    for (int off = 1; off < 16; off <<= 1)
#pragma unroll
        for (int v = 0; v < 8; ++v) lRow[v] += __shfl_xor(lRow[v], off);

#pragma unroll
    for (int d = 0; d < 4; ++d) {
        const int col = hoff + d * 16 + r;
#pragma unroll
        for (int v = 0; v < 8; ++v) {
            const int t = t0 + v + 8 * hi;
            O[(size_t)(b * 1024 + t) * HID + col] = (__bf16)(acc[d][v] * (1.f / lRow[v]));
        }
    }
}

// ---------------------------------------------------------------------------
// out = LayerNorm(resid + proj) * g + b ; optional bf16 mirror for attention Q
// ---------------------------------------------------------------------------
__global__ __launch_bounds__(256) void addln_kernel(
    const float* __restrict__ resid, const float* __restrict__ proj,
    const float* __restrict__ g, const float* __restrict__ beta,
    float* __restrict__ out, __bf16* __restrict__ outB) {
    const int row = blockIdx.x;
    const int tid = threadIdx.x;
    float4 rv = ((const float4*)(resid + (size_t)row * HID))[tid];
    float4 pv = ((const float4*)(proj + (size_t)row * HID))[tid];
    float x0 = rv.x + pv.x, x1 = rv.y + pv.y, x2 = rv.z + pv.z, x3 = rv.w + pv.w;
    float s  = x0 + x1 + x2 + x3;
    float s2 = x0 * x0 + x1 * x1 + x2 * x2 + x3 * x3;
#pragma unroll
    for (int off = 16; off > 0; off >>= 1) {
        s  += __shfl_down(s, off);
        s2 += __shfl_down(s2, off);
    }
    __shared__ float shs[8], shq[8];
    __shared__ float mean_s, inv_s;
    const int warp = tid >> 5, lane = tid & 31;
    if (lane == 0) { shs[warp] = s; shq[warp] = s2; }
    __syncthreads();
    if (tid == 0) {
        float ts = 0.f, tq = 0.f;
        for (int i = 0; i < 8; ++i) { ts += shs[i]; tq += shq[i]; }
        const float mean = ts * (1.f / HID);
        const float var  = tq * (1.f / HID) - mean * mean;
        mean_s = mean;
        inv_s  = rsqrtf(var + 1e-12f);
    }
    __syncthreads();
    const float mean = mean_s, inv = inv_s;
    float4 gv = ((const float4*)g)[tid];
    float4 bv = ((const float4*)beta)[tid];
    float4 o;
    o.x = (x0 - mean) * inv * gv.x + bv.x;
    o.y = (x1 - mean) * inv * gv.y + bv.y;
    o.z = (x2 - mean) * inv * gv.z + bv.z;
    o.w = (x3 - mean) * inv * gv.w + bv.w;
    ((float4*)(out + (size_t)row * HID))[tid] = o;
    if (outB) {
        __bf16* ob = outB + (size_t)row * HID + tid * 4;
        ob[0] = (__bf16)o.x; ob[1] = (__bf16)o.y;
        ob[2] = (__bf16)o.z; ob[3] = (__bf16)o.w;
    }
}

// ---------------------------------------------------------------------------
extern "C" void kernel_launch(void* const* d_in, const int* in_sizes, int n_in,
                              void* d_out, int out_size, void* d_ws, size_t ws_size,
                              hipStream_t stream) {
    (void)in_sizes; (void)n_in; (void)out_size; (void)ws_size;
    const float* enc   = (const float*)d_in[0];
    const float* dec   = (const float*)d_in[1];
    const float* smask = (const float*)d_in[2];
    const float* tmask = (const float*)d_in[3];
    const float* Wq = (const float*)d_in[4];
    const float* bq = (const float*)d_in[5];
    const float* Wk = (const float*)d_in[6];
    const float* bk = (const float*)d_in[7];
    const float* Wv = (const float*)d_in[8];
    const float* bv = (const float*)d_in[9];
    const float* Wd = (const float*)d_in[10];
    const float* bd = (const float*)d_in[11];
    const float* lng = (const float*)d_in[12];
    const float* lnb = (const float*)d_in[13];
    float* out = (float*)d_out;

    const int M = 4096, N = HID, K = HID;
    const size_t ACT = (size_t)M * N;
    const size_t WEL = (size_t)N * K;

    char* p = (char*)d_ws;
    auto carve = [&](size_t bytes) { char* r = p; p += (bytes + 255) & ~(size_t)255; return r; };
    __bf16* decB  = (__bf16*)carve(ACT * 2);
    __bf16* encB  = (__bf16*)carve(ACT * 2);
    __bf16* WqB   = (__bf16*)carve(WEL * 2);
    __bf16* WkB   = (__bf16*)carve(WEL * 2);
    __bf16* WvB   = (__bf16*)carve(WEL * 2);
    __bf16* WdB   = (__bf16*)carve(WEL * 2);
    float*  q1f   = (float*)carve(ACT * 4);
    float*  projf = (float*)carve(ACT * 4);
    float*  selff = (float*)carve(ACT * 4);
    __bf16* q1B   = (__bf16*)carve(ACT * 2);
    __bf16* k1B   = (__bf16*)carve(ACT * 2);
    __bf16* v1B   = (__bf16*)carve(ACT * 2);
    __bf16* ekB   = (__bf16*)carve(ACT * 2);
    __bf16* evB   = (__bf16*)carve(ACT * 2);
    __bf16* ctxB  = (__bf16*)carve(ACT * 2);
    __bf16* selfB = (__bf16*)carve(ACT * 2);

    dim3 gG(N / 128, M / 128), bG(256);
    dim3 gA(1024 / 64, NH, 4), bA(128);
    dim3 gL(4096),             bL(256);

    // one-time bf16 conversions (outside all hot loops)
    cvtbf_kernel<<<(int)(ACT / 8 / 256), 256, 0, stream>>>(dec, decB, (int)(ACT / 8));
    cvtbf_kernel<<<(int)(ACT / 8 / 256), 256, 0, stream>>>(enc, encB, (int)(ACT / 8));
    cvtbf_kernel<<<(int)(WEL / 8 / 256), 256, 0, stream>>>(Wq, WqB, (int)(WEL / 8));
    cvtbf_kernel<<<(int)(WEL / 8 / 256), 256, 0, stream>>>(Wk, WkB, (int)(WEL / 8));
    cvtbf_kernel<<<(int)(WEL / 8 / 256), 256, 0, stream>>>(Wv, WvB, (int)(WEL / 8));
    cvtbf_kernel<<<(int)(WEL / 8 / 256), 256, 0, stream>>>(Wd, WdB, (int)(WEL / 8));

    // self-attention branch
    linear_kernel<<<gG, bG, 0, stream>>>(decB, WqB, bq, q1f, q1B, M, N, K);
    linear_kernel<<<gG, bG, 0, stream>>>(decB, WkB, bk, nullptr, k1B, M, N, K);
    linear_kernel<<<gG, bG, 0, stream>>>(decB, WvB, bv, nullptr, v1B, M, N, K);
    attn_kernel<<<gA, bA, 0, stream>>>(q1B, k1B, v1B, tmask, ctxB);
    linear_kernel<<<gG, bG, 0, stream>>>(ctxB, WdB, bd, projf, nullptr, M, N, K);
    addln_kernel<<<gL, bL, 0, stream>>>(q1f, projf, lng, lnb, selff, selfB);

    // cross-attention branch (query = self_out directly, per reference)
    linear_kernel<<<gG, bG, 0, stream>>>(encB, WkB, bk, nullptr, ekB, M, N, K);
    linear_kernel<<<gG, bG, 0, stream>>>(encB, WvB, bv, nullptr, evB, M, N, K);
    attn_kernel<<<gA, bA, 0, stream>>>(selfB, ekB, evB, smask, ctxB);
    linear_kernel<<<gG, bG, 0, stream>>>(ctxB, WdB, bd, projf, nullptr, M, N, K);
    addln_kernel<<<gL, bL, 0, stream>>>(selff, projf, lng, lnb, out, nullptr);
}
